// MultiHeadAttention_26199300505773
// MI455X (gfx1250) — matrix operand order repaired and checked by reading, not compiled, not run on hardware
//
#include <hip/hip_runtime.h>
#include <math.h>

// ---------------------------------------------------------------------------
// MHA forward for MI455X (gfx1250, wave32, WMMA).
// B=4, S=2048, E=2048, H=16, D=128.
// f16 WMMA (f32 accumulate) GEMMs + fused flash attention.
// LDS tiles double-buffered via GLOBAL_LOAD_ASYNC_TO_LDS (ASYNCcnt).
// ---------------------------------------------------------------------------

typedef __attribute__((ext_vector_type(16))) _Float16 v16h;
typedef __attribute__((ext_vector_type(8)))  _Float16 v8h;
typedef __attribute__((ext_vector_type(4)))  _Float16 v4h;
typedef __attribute__((ext_vector_type(8)))  float    v8f;

constexpr int Bb = 4;
constexpr int Ss = 2048;
constexpr int Ee = 2048;
constexpr int Hh = 16;
constexpr int Dd = 128;
constexpr int BS = Bb * Ss;           // 8192 tokens
constexpr float EPS = 1e-6f;

// ---------------------------------------------------------------------------
// WMMA helper: D = A(16x32 f16) * B(32x16 f16) + C(16x16 f32)
// ---------------------------------------------------------------------------
__device__ __forceinline__ v8f wmma16(v16h a, v16h b, v8f c) {
  return __builtin_amdgcn_wmma_f32_16x16x32_f16(
      /*neg_a=*/false, a, /*neg_b=*/false, b,
      /*c_mod=*/(short)0, c, /*reuse_a=*/false, /*reuse_b=*/false);
}

// Async cache->LDS copy of 16 bytes per lane (tracked by ASYNCcnt).
// vdst = per-lane LDS byte address (low 32 bits of generic shared pointer),
// vaddr = 64-bit global address.
__device__ __forceinline__ void async_copy_b128(const _Float16* g, _Float16* l) {
  unsigned lds_addr = (unsigned)(size_t)(void*)l;
  asm volatile("global_load_async_to_lds_b128 %0, %1, off"
               :: "v"(lds_addr), "v"(g) : "memory");
}
__device__ __forceinline__ void wait_async0() {
  asm volatile("s_wait_asynccnt 0" ::: "memory");
}

// A-matrix fragment (16x32 f16), row-major source with leading dim `ld`.
// ISA 7.12.2: lane<16 holds K = {0..7, 16..23}; lane>=16 holds K = {8..15, 24..31}.
__device__ __forceinline__ v16h load_frag_a(const _Float16* __restrict__ base,
                                            int ld, int row, int k0, int half) {
  const _Float16* p = base + (size_t)row * ld + k0 + half * 8;
  v8h lo = *(const v8h*)(p);        // K offsets half*8 + 0..7
  v8h hi = *(const v8h*)(p + 16);   // K offsets 16 + half*8 + 0..7
  v16h f;
#pragma unroll
  for (int i = 0; i < 8; ++i) { f[i] = lo[i]; f[i + 8] = hi[i]; }
  return f;
}

// B-matrix fragment (32x16 f16). Source row-major [N][K] (A*B^T style:
// element (k,n) = src[n*ld + k]). Same K order as A: lane<16 holds
// K = {0..7, 16..23}; lane>=16 holds K = {8..15, 24..31}, N = lane&15.
__device__ __forceinline__ v16h load_frag_b(const _Float16* __restrict__ base,
                                            int ld, int col, int k0, int half) {
  const _Float16* p = base + (size_t)col * ld + k0 + half * 8;
  v8h lo = *(const v8h*)(p);        // K offsets half*8 + 0..7
  v8h hi = *(const v8h*)(p + 16);   // K offsets 16 + half*8 + 0..7
  v16h f;
#pragma unroll
  for (int i = 0; i < 8; ++i) { f[i] = lo[i]; f[i + 8] = hi[i]; }
  return f;
}

// half-wave (16-lane group) reductions — wave32 native
__device__ __forceinline__ float redmax16(float v) {
  v = fmaxf(v, __shfl_xor(v, 1, 32));
  v = fmaxf(v, __shfl_xor(v, 2, 32));
  v = fmaxf(v, __shfl_xor(v, 4, 32));
  v = fmaxf(v, __shfl_xor(v, 8, 32));
  return v;
}
__device__ __forceinline__ float redsum16(float v) {
  v += __shfl_xor(v, 1, 32);
  v += __shfl_xor(v, 2, 32);
  v += __shfl_xor(v, 4, 32);
  v += __shfl_xor(v, 8, 32);
  return v;
}

// ---------------------------------------------------------------------------
// f32 -> f16 convert (vectorized, grid-stride)
// ---------------------------------------------------------------------------
__global__ void k_cvt(const float* __restrict__ in, _Float16* __restrict__ out, int n) {
  int i = (blockIdx.x * blockDim.x + threadIdx.x) * 4;
  int stride = gridDim.x * blockDim.x * 4;
  for (; i < n; i += stride) {
    float4 v = *(const float4*)(in + i);
    v4h o;
    o[0] = (_Float16)v.x; o[1] = (_Float16)v.y;
    o[2] = (_Float16)v.z; o[3] = (_Float16)v.w;
    *(v4h*)(out + i) = o;
  }
}

// ---------------------------------------------------------------------------
// GEMM: C[M,N] = A[M,K] * W[N,K]^T   (x @ W.T), f16 in, f32 accumulate.
// Block tile 128(M) x 128(N): 8 waves, wave w owns rows [m0+16w, +16) and all
// 128 N. 128x64 B-chunks stream into a 2-stage LDS buffer via async-to-LDS.
// grid = (M/128, N/128), block = 256.
// ---------------------------------------------------------------------------
__global__ __launch_bounds__(256) void k_gemm(const _Float16* __restrict__ A,
                                              const _Float16* __restrict__ W,
                                              _Float16* __restrict__ Ch,
                                              float* __restrict__ Cf,
                                              int M, int N, int K) {
  __shared__ __align__(16) _Float16 bstage[2][128 * 64];   // 2 x 16KB [n][k]

  const int tid = threadIdx.x;
  const int wv = tid >> 5;
  const int lane = tid & 31;
  const int lane15 = lane & 15;
  const int half = lane >> 4;
  const int m0 = blockIdx.x * 128 + wv * 16;
  const int n0 = blockIdx.y * 128;

  // issue this thread's 4 x b128 async copies of a 128x64 B-chunk
  auto issue_b = [&](int stage, int k0) {
#pragma unroll
    for (int i = 0; i < 4; ++i) {
      int c = tid * 4 + i;                  // [0, 1024) 16B chunks
      int row = c >> 3, off = (c & 7) * 8;
      async_copy_b128(W + (size_t)(n0 + row) * K + k0 + off,
                      &bstage[stage][row * 64 + off]);
    }
  };

  v8f acc[8];
#pragma unroll
  for (int t = 0; t < 8; ++t) acc[t] = (v8f){};

  issue_b(0, 0);

#pragma unroll 1
  for (int k0 = 0; k0 < K; k0 += 64) {
    const int s = (k0 >> 6) & 1;
    wait_async0();              // my async copies into stage s have landed
    __syncthreads();            // everyone's landed; stage s^1 reads complete
    if (k0 + 64 < K) issue_b(s ^ 1, k0 + 64);   // overlap with compute

    const _Float16* bs = &bstage[s][0];
#pragma unroll
    for (int kk = 0; kk < 2; ++kk) {
      v16h af = load_frag_a(A, K, m0 + lane15, k0 + kk * 32, half);
#pragma unroll
      for (int t = 0; t < 8; ++t) {
        v16h bf = load_frag_b(bs, 64, t * 16 + lane15, kk * 32, half);
        acc[t] = wmma16(af, bf, acc[t]);
      }
    }
  }

  // C layout: VGPR r -> row m0 + r + half*8, col = n0 + t*16 + lane15
#pragma unroll
  for (int t = 0; t < 8; ++t) {
#pragma unroll
    for (int r = 0; r < 8; ++r) {
      size_t ci = (size_t)(m0 + r + half * 8) * N + n0 + t * 16 + lane15;
      if (Ch) Ch[ci] = (_Float16)acc[t][r];
      else    Cf[ci] = acc[t][r];
    }
  }
}

// ---------------------------------------------------------------------------
// RoPE + RMSNorm for Q,K; transpose V.
// One wave per (token, head) row of 128 elements; lane covers d = 4*lane..+3.
// qraw/kraw/vraw: [token, E] f16.  Qh/Kh: [b,h,s,d].  Vt: [b,h,d,s].
// ---------------------------------------------------------------------------
__device__ __forceinline__ void rope_norm_row(const _Float16* __restrict__ src,
                                              _Float16* __restrict__ dst,
                                              float4 cv, float4 sv, int lane) {
  v4h qv = *(const v4h*)src;
  float q[4], rot[4], e[4];
#pragma unroll
  for (int i = 0; i < 4; ++i) q[i] = (float)qv[i];
#pragma unroll
  for (int i = 0; i < 4; ++i) {
    float p = __shfl_xor(q[i], 16, 32);       // partner holds d +/- 64
    rot[i] = (lane < 16) ? -p : p;            // rotate_half
  }
  e[0] = q[0] * cv.x + rot[0] * sv.x;
  e[1] = q[1] * cv.y + rot[1] * sv.y;
  e[2] = q[2] * cv.z + rot[2] * sv.z;
  e[3] = q[3] * cv.w + rot[3] * sv.w;
  float ss = e[0]*e[0] + e[1]*e[1] + e[2]*e[2] + e[3]*e[3];
  ss += __shfl_xor(ss, 1, 32);
  ss += __shfl_xor(ss, 2, 32);
  ss += __shfl_xor(ss, 4, 32);
  ss += __shfl_xor(ss, 8, 32);
  ss += __shfl_xor(ss, 16, 32);
  float sc = rsqrtf(ss * (1.0f / (float)Dd) + EPS);
  v4h ov;
#pragma unroll
  for (int i = 0; i < 4; ++i) ov[i] = (_Float16)(e[i] * sc);
  *(v4h*)dst = ov;
}

__global__ __launch_bounds__(256) void k_rope_norm(
    const _Float16* __restrict__ qraw, const _Float16* __restrict__ kraw,
    const _Float16* __restrict__ vraw,
    const float* __restrict__ cosp, const float* __restrict__ sinp,
    _Float16* __restrict__ Qh, _Float16* __restrict__ Kh,
    _Float16* __restrict__ Vt) {
  const int wv = threadIdx.x >> 5;
  const int lane = threadIdx.x & 31;
  const int w = blockIdx.x * 8 + wv;          // [0, B*S*H)
  const int h = w % Hh;
  const int token = w / Hh;                   // b*S + s
  const int s = token % Ss;
  const int b = token / Ss;
  const int d0 = lane * 4;

  const size_t src = (size_t)token * Ee + (size_t)h * Dd + d0;
  const size_t dst = (((size_t)(b * Hh + h) * Ss) + s) * Dd + d0;

  float4 cv = *(const float4*)(cosp + (size_t)s * Dd + d0);
  float4 sv = *(const float4*)(sinp + (size_t)s * Dd + d0);

  rope_norm_row(qraw + src, Qh + dst, cv, sv, lane);
  rope_norm_row(kraw + src, Kh + dst, cv, sv, lane);

  // V: no rope/norm, just transpose into [b,h,d,s]
  v4h vv = *(const v4h*)(vraw + src);
  size_t vb = ((size_t)(b * Hh + h) * Dd + d0) * Ss + s;
#pragma unroll
  for (int i = 0; i < 4; ++i) Vt[vb + (size_t)i * Ss] = vv[i];
}

// ---------------------------------------------------------------------------
// Fused flash attention. Block = 8 waves = 8 query tiles (128 queries) of the
// SAME (b,h). Per 32-key chunk, the K-chunk (32x128) and V-chunk (128x32 of
// Vt) stream into 2-stage LDS buffers via async-to-LDS, shared by all waves.
// Qh/Kh: [b,h,s,d]; Vt: [b,h,d,s]; Y: [token, E] f16.
// ---------------------------------------------------------------------------
__global__ __launch_bounds__(256) void k_attn(const _Float16* __restrict__ Qh,
                                              const _Float16* __restrict__ Kh,
                                              const _Float16* __restrict__ Vt,
                                              _Float16* __restrict__ Y) {
  __shared__ __align__(16) _Float16 kst[2][32 * 128];   // [key][d]  2 x 8KB
  __shared__ __align__(16) _Float16 vst[2][128 * 32];   // [d][key]  2 x 8KB
  __shared__ __align__(16) _Float16 pbuf[8][16 * 32];   // per-wave P, 8KB

  const int tid = threadIdx.x;
  const int wv = tid >> 5;
  const int lane = tid & 31;
  const int lane15 = lane & 15;
  const int half = lane >> 4;

  const int qblock = blockIdx.x % (Ss / 128);  // 16 query-blocks per (b,h)
  const int bh = blockIdx.x / (Ss / 128);      // b*H + h
  const int b = bh / Hh, h = bh % Hh;

  const _Float16* Qp = Qh + ((size_t)bh * Ss + qblock * 128 + wv * 16) * Dd;
  const _Float16* Kp = Kh + (size_t)bh * Ss * Dd;
  const _Float16* Vp = Vt + (size_t)bh * Dd * Ss;       // [d][s]

  // issue this thread's async copies for one 32-key chunk (2 K + 2 V b128s)
  auto issue_kv = [&](int stage, int kc) {
#pragma unroll
    for (int i = 0; i < 2; ++i) {
      int c = tid * 2 + i;                    // [0, 512) 16B chunks
      { int row = c >> 4, off = (c & 15) * 8; // K: 16 chunks per 256B row
        async_copy_b128(Kp + (size_t)(kc + row) * Dd + off,
                        &kst[stage][row * 128 + off]); }
      { int row = c >> 2, off = (c & 3) * 8;  // V: 4 chunks per 64B row
        async_copy_b128(Vp + (size_t)row * Ss + kc + off,
                        &vst[stage][row * 32 + off]); }
    }
  };

  // Q tile as 4 A-fragments over D = 4 * 32
  v16h qf[4];
#pragma unroll
  for (int j = 0; j < 4; ++j) qf[j] = load_frag_a(Qp, Dd, lane15, j * 32, half);

  v8f o[8];
#pragma unroll
  for (int t = 0; t < 8; ++t) o[t] = (v8f){};
  float m_i[8], l_i[8];
#pragma unroll
  for (int r = 0; r < 8; ++r) { m_i[r] = -1e30f; l_i[r] = 0.0f; }

  const float sc = 0.08838834764831845f;      // 1/sqrt(128)
  _Float16* pb = &pbuf[wv][0];

  issue_kv(0, 0);

#pragma unroll 1
  for (int kc = 0; kc < Ss; kc += 32) {
    const int s = (kc >> 5) & 1;
    wait_async0();              // my async copies into stage s have landed
    __syncthreads();            // all landed; stage s^1 reads complete
    if (kc + 32 < Ss) issue_kv(s ^ 1, kc + 32);

    const _Float16* ks = &kst[s][0];
    const _Float16* vs = &vst[s][0];

    // scores for two 16-key sub-tiles (from LDS)
    v8f s0 = {}, s1 = {};
#pragma unroll
    for (int j = 0; j < 4; ++j) {
      v16h b0 = load_frag_b(ks, Dd, lane15, j * 32, half);
      s0 = wmma16(qf[j], b0, s0);
      v16h b1 = load_frag_b(ks + 16 * Dd, Dd, lane15, j * 32, half);
      s1 = wmma16(qf[j], b1, s1);
    }

    // online softmax update (row r of this half-wave = r + half*8)
    float alpha[8];
#pragma unroll
    for (int r = 0; r < 8; ++r) {
      s0[r] *= sc;
      s1[r] *= sc;
      float mx = redmax16(fmaxf(s0[r], s1[r]));
      float mn = fmaxf(m_i[r], mx);
      alpha[r] = __expf(m_i[r] - mn);
      m_i[r] = mn;
      s0[r] = __expf(s0[r] - mn);
      s1[r] = __expf(s1[r] - mn);
      float rs = redsum16(s0[r] + s1[r]);
      l_i[r] = l_i[r] * alpha[r] + rs;
    }
#pragma unroll
    for (int t = 0; t < 8; ++t)
#pragma unroll
      for (int r = 0; r < 8; ++r) o[t][r] *= alpha[r];

    // P tile (16 q x 32 keys) C-layout -> per-wave LDS region (row-major f16)
#pragma unroll
    for (int r = 0; r < 8; ++r) {
      int row = r + half * 8;
      pb[row * 32 + lane15]      = (_Float16)s0[r];
      pb[row * 32 + 16 + lane15] = (_Float16)s1[r];
    }
    // same-wave cross-lane LDS store->load dependency: drain DScnt
    asm volatile("s_wait_dscnt 0" ::: "memory");

    // reload P as A-fragment (16x32)
    v16h pf = load_frag_a(pb, 32, lane15, 0, half);

    // O += P * V  (V fragments from staged Vt chunk, contiguous along keys)
#pragma unroll
    for (int t = 0; t < 8; ++t) {
      v16h vf = load_frag_b(vs, 32, t * 16 + lane15, 0, half);
      o[t] = wmma16(pf, vf, o[t]);
    }
  }

#pragma unroll
  for (int r = 0; r < 8; ++r) l_i[r] = 1.0f / l_i[r];

  // write Y[token, h*D + d]
#pragma unroll
  for (int t = 0; t < 8; ++t) {
#pragma unroll
    for (int r = 0; r < 8; ++r) {
      int q = qblock * 128 + wv * 16 + r + half * 8;
      size_t yi = (size_t)(b * Ss + q) * Ee + h * Dd + t * 16 + lane15;
      Y[yi] = (_Float16)(o[t][r] * l_i[r]);
    }
  }
}

// ---------------------------------------------------------------------------
// launch
// ---------------------------------------------------------------------------
extern "C" void kernel_launch(void* const* d_in, const int* in_sizes, int n_in,
                              void* d_out, int out_size, void* d_ws, size_t ws_size,
                              hipStream_t stream) {
  (void)in_sizes; (void)n_in; (void)out_size; (void)ws_size;

  const float* x    = (const float*)d_in[0];
  const float* cosp = (const float*)d_in[1];
  const float* sinp = (const float*)d_in[2];
  const float* Wq   = (const float*)d_in[3];
  const float* Wk   = (const float*)d_in[4];
  const float* Wv   = (const float*)d_in[5];
  const float* Wo   = (const float*)d_in[6];

  const size_t MB = 1024ull * 1024ull;
  char* p = (char*)d_ws;
  _Float16* xh   = (_Float16*)(p + 0 * MB);     // 32MB; reused as Qh later
  _Float16* wqh  = (_Float16*)(p + 32 * MB);    // 8MB each
  _Float16* wkh  = (_Float16*)(p + 40 * MB);
  _Float16* wvh  = (_Float16*)(p + 48 * MB);
  _Float16* woh  = (_Float16*)(p + 56 * MB);
  _Float16* qraw = (_Float16*)(p + 64 * MB);    // 32MB; reused as Y later
  _Float16* kraw = (_Float16*)(p + 96 * MB);
  _Float16* vraw = (_Float16*)(p + 128 * MB);
  _Float16* Kh   = (_Float16*)(p + 160 * MB);
  _Float16* Vt   = (_Float16*)(p + 192 * MB);   // ends at 224MB
  _Float16* Qh = xh;                            // alias (xh dead after GEMMs)
  _Float16* Y  = qraw;                          // alias (qraw dead after rope)

  // 1) f32 -> f16 converts
  k_cvt<<<8192, 256, 0, stream>>>(x,  xh,  BS * Ee);
  k_cvt<<<2048, 256, 0, stream>>>(Wq, wqh, Ee * Ee);
  k_cvt<<<2048, 256, 0, stream>>>(Wk, wkh, Ee * Ee);
  k_cvt<<<2048, 256, 0, stream>>>(Wv, wvh, Ee * Ee);
  k_cvt<<<2048, 256, 0, stream>>>(Wo, woh, Ee * Ee);

  // 2) QKV projections: [8192,2048] @ [2048,2048]^T, f16 out
  dim3 gg(BS / 128, Ee / 128);                  // (64, 16)
  k_gemm<<<gg, 256, 0, stream>>>(xh, wqh, qraw, nullptr, BS, Ee, Ee);
  k_gemm<<<gg, 256, 0, stream>>>(xh, wkh, kraw, nullptr, BS, Ee, Ee);
  k_gemm<<<gg, 256, 0, stream>>>(xh, wvh, vraw, nullptr, BS, Ee, Ee);

  // 3) RoPE + RMSNorm (Q,K) + V transpose; one wave per (token, head)
  k_rope_norm<<<(Bb * Ss * Hh) / 8, 256, 0, stream>>>(
      qraw, kraw, vraw, cosp, sinp, Qh, Kh, Vt);

  // 4) fused flash attention; block = 8 query tiles of one (b,h)
  k_attn<<<Bb * Hh * (Ss / 128), 256, 0, stream>>>(Qh, Kh, Vt, Y);

  // 5) output projection -> fp32 d_out
  k_gemm<<<gg, 256, 0, stream>>>(Y, woh, nullptr, (float*)d_out, BS, Ee, Ee);
}
